// NeuralDMD_Stage1_23416161698148
// MI455X (gfx1250) — compile-verified, matrix operand order checked by reading, not run on hardware
//
#include <hip/hip_runtime.h>
#include <math.h>

// ---------------------------------------------------------------------------
// Types for CDNA5 WMMA
// ---------------------------------------------------------------------------
typedef __attribute__((ext_vector_type(16))) __bf16 v16bf;
typedef __attribute__((ext_vector_type(8)))  float  v8f;
typedef __attribute__((ext_vector_type(2)))  float  v2f;

union BFrag {
    v16bf v;
    unsigned short h[16];
    uint4 q[2];
};

__device__ __forceinline__ unsigned short f2bf(float f) {
    union { float f; unsigned u; } x; x.f = f;
    unsigned u = x.u;
    return (unsigned short)((u + 0x7FFFu + ((u >> 16) & 1u)) >> 16);
}

// ---------------------------------------------------------------------------
// Pack W (KxN f32, row-major) -> Wt (NxK bf16, row-major)   (B-fragment friendly)
// ---------------------------------------------------------------------------
__global__ void k_pack_wt(const float* __restrict__ W, unsigned short* __restrict__ Wt,
                          int K, int N) {
    int idx = blockIdx.x * blockDim.x + threadIdx.x;
    if (idx >= K * N) return;
    int n = idx / K, k = idx - n * K;
    Wt[idx] = f2bf(W[(size_t)k * N + n]);
}

// ---------------------------------------------------------------------------
// MLP layer: C(Mx512) = act(A(MxK) @ Wt^T + bias)
//   A is bf16 (ushort) row-major, or f32 row-major if a_is_f32 (on-the-fly cvt)
//   Wt is bf16 N x K row-major (pre-transposed weights), N = 512
//   store_f32 == 0: relu, write bf16 to outb ; else: no relu, write f32 to outf
// Block: 256 threads = 8 waves; wave tile 32x64; block tile 128x128.
// Uses v_wmma_f32_16x16x32_bf16.
// ---------------------------------------------------------------------------
__global__ __launch_bounds__(256)
void k_layer_gemm(const void* __restrict__ Av, int a_is_f32,
                  const unsigned short* __restrict__ Wt,
                  const float* __restrict__ bias,
                  unsigned short* __restrict__ outb, float* __restrict__ outf,
                  int store_f32, int M, int K) {
    const int lane = threadIdx.x & 31;
    const int w    = threadIdx.x >> 5;
    const int wm   = w >> 1;              // 0..3 -> M offset 32*wm
    const int wn   = w & 1;               // 0..1 -> N offset 64*wn
    const int bm   = blockIdx.y * 128;
    const int bn   = blockIdx.x * 128;
    const int rl   = lane & 15;
    const int kh   = (lane >> 4) * 8;     // 0 or 8 (bf16 A/B fragment K phase)

    v8f acc[2][4];
    #pragma unroll
    for (int t = 0; t < 2; ++t)
        #pragma unroll
        for (int u = 0; u < 4; ++u)
            #pragma unroll
            for (int i = 0; i < 8; ++i) acc[t][u][i] = 0.0f;

    const unsigned short* Ab = (const unsigned short*)Av;
    const float*          Af = (const float*)Av;

    for (int k = 0; k < K; k += 32) {
        BFrag a[2], b[4];
        #pragma unroll
        for (int t = 0; t < 2; ++t) {
            int row = bm + wm * 32 + t * 16 + rl;
            if (a_is_f32) {
                const float* p = Af + (size_t)row * K + k + kh;
                float4 f0 = *(const float4*)(p);
                float4 f1 = *(const float4*)(p + 4);
                float4 f2 = *(const float4*)(p + 16);
                float4 f3 = *(const float4*)(p + 20);
                a[t].h[0]  = f2bf(f0.x); a[t].h[1]  = f2bf(f0.y);
                a[t].h[2]  = f2bf(f0.z); a[t].h[3]  = f2bf(f0.w);
                a[t].h[4]  = f2bf(f1.x); a[t].h[5]  = f2bf(f1.y);
                a[t].h[6]  = f2bf(f1.z); a[t].h[7]  = f2bf(f1.w);
                a[t].h[8]  = f2bf(f2.x); a[t].h[9]  = f2bf(f2.y);
                a[t].h[10] = f2bf(f2.z); a[t].h[11] = f2bf(f2.w);
                a[t].h[12] = f2bf(f3.x); a[t].h[13] = f2bf(f3.y);
                a[t].h[14] = f2bf(f3.z); a[t].h[15] = f2bf(f3.w);
            } else {
                const unsigned short* p = Ab + (size_t)row * K + k + kh;
                a[t].q[0] = *(const uint4*)(p);
                a[t].q[1] = *(const uint4*)(p + 16);
            }
        }
        #pragma unroll
        for (int u = 0; u < 4; ++u) {
            int col = bn + wn * 64 + u * 16 + rl;
            const unsigned short* p = Wt + (size_t)col * K + k + kh;
            b[u].q[0] = *(const uint4*)(p);
            b[u].q[1] = *(const uint4*)(p + 16);
        }
        #pragma unroll
        for (int t = 0; t < 2; ++t)
            #pragma unroll
            for (int u = 0; u < 4; ++u)
                acc[t][u] = __builtin_amdgcn_wmma_f32_16x16x32_bf16(
                    false, a[t].v, false, b[u].v, (short)0, acc[t][u], false, false);
    }

    const int hr = (lane >> 4) * 8;       // C/D layout: VGPR i -> row hr+i
    #pragma unroll
    for (int t = 0; t < 2; ++t)
        #pragma unroll
        for (int u = 0; u < 4; ++u) {
            int col = bn + wn * 64 + u * 16 + rl;
            float bv = bias[col];
            #pragma unroll
            for (int i = 0; i < 8; ++i) {
                int row = bm + wm * 32 + t * 16 + hr + i;
                float vv = acc[t][u][i] + bv;
                if (!store_f32) {
                    vv = fmaxf(vv, 0.0f);
                    outb[(size_t)row * 512 + col] = f2bf(vv);
                } else {
                    outf[(size_t)row * 512 + col] = vv;
                }
            }
        }
}

// ---------------------------------------------------------------------------
// f32 GEMM, C(MxN) = A(MxK) @ B(KxN), row-major, N multiple of 128, K mult of 4.
// Wave computes one 16x16 tile via v_wmma_f32_16x16x4_f32; M bounds-checked.
// ---------------------------------------------------------------------------
__global__ __launch_bounds__(256)
void k_gemm_f32_nn(const float* __restrict__ A, const float* __restrict__ B,
                   float* __restrict__ C, int M, int N, int K, int ldb) {
    const int lane = threadIdx.x & 31;
    const int w    = threadIdx.x >> 5;
    const int bn   = blockIdx.x * 128 + w * 16;
    const int bm   = blockIdx.y * 16;
    const int m    = lane & 15;
    const int k0   = (lane >> 4) * 2;     // f32 A/B fragment K phase
    const int row  = bm + m;
    const int col  = bn + m;

    v8f acc;
    #pragma unroll
    for (int i = 0; i < 8; ++i) acc[i] = 0.0f;

    for (int k = 0; k < K; k += 4) {
        v2f a, b;
        if (row < M) {
            a.x = A[(size_t)row * K + k + k0];
            a.y = A[(size_t)row * K + k + k0 + 1];
        } else {
            a.x = 0.0f; a.y = 0.0f;
        }
        b.x = B[(size_t)(k + k0)     * ldb + col];
        b.y = B[(size_t)(k + k0 + 1) * ldb + col];
        acc = __builtin_amdgcn_wmma_f32_16x16x4_f32(
            false, a, false, b, (short)0, acc, false, false);
    }
    const int hr = (lane >> 4) * 8;
    #pragma unroll
    for (int i = 0; i < 8; ++i) {
        int r = bm + hr + i;
        if (r < M) C[(size_t)r * N + col] = acc[i];
    }
}

// ---------------------------------------------------------------------------
// f32 "TN" GEMM over samples: C(512x512) = A1^T @ A2  (A1,A2: K x 512 row-major)
// Writes into C with row stride ldc at column offset colOff; adds ridge on diag.
// ---------------------------------------------------------------------------
__global__ __launch_bounds__(256)
void k_gemm_f32_tn(const float* __restrict__ A1, const float* __restrict__ A2,
                   float* __restrict__ C, int K, int ldc, int colOff, float ridge) {
    const int lane = threadIdx.x & 31;
    const int w    = threadIdx.x >> 5;
    const int bn   = blockIdx.x * 128 + w * 16;
    const int bm   = blockIdx.y * 16;
    const int m    = lane & 15;
    const int k0   = (lane >> 4) * 2;

    v8f acc;
    #pragma unroll
    for (int i = 0; i < 8; ++i) acc[i] = 0.0f;

    for (int k = 0; k < K; k += 4) {
        v2f a, b;
        a.x = A1[(size_t)(k + k0)     * 512 + bm + m];
        a.y = A1[(size_t)(k + k0 + 1) * 512 + bm + m];
        b.x = A2[(size_t)(k + k0)     * 512 + bn + m];
        b.y = A2[(size_t)(k + k0 + 1) * 512 + bn + m];
        acc = __builtin_amdgcn_wmma_f32_16x16x4_f32(
            false, a, false, b, (short)0, acc, false, false);
    }
    const int hr = (lane >> 4) * 8;
    #pragma unroll
    for (int i = 0; i < 8; ++i) {
        int r = bm + hr + i;
        int c = bn + m;
        float v = acc[i];
        if (r == c) v += ridge;
        C[(size_t)r * ldc + colOff + c] = v;
    }
}

// ---------------------------------------------------------------------------
// Gauss-Jordan elimination step on Gaug (512 x 1024), pivot p.
// Only columns c > p are updated -> no intra-launch races; diagonal + right
// half stay exact.  One launch per pivot (graph nodes amortize overhead).
// ---------------------------------------------------------------------------
__global__ void k_gj_elim(float* __restrict__ Gaug, int p) {
    int idx = blockIdx.x * blockDim.x + threadIdx.x;   // 512*1024 threads
    int r = idx >> 10;
    int c = idx & 1023;
    if (r == p || c <= p) return;
    float piv = Gaug[(size_t)p * 1024 + p];
    float f   = Gaug[(size_t)r * 1024 + p] / piv;
    Gaug[(size_t)r * 1024 + c] -= f * Gaug[(size_t)p * 1024 + c];
}

// Finalize: X = D^{-1} * RightHalf solves G X = A2^T, so X = K^T.
// Write X (= N0 for the doubling recursion) and K = X^T into d_out+1.
__global__ void k_gj_final(const float* __restrict__ Gaug,
                           float* __restrict__ X, float* __restrict__ Kout) {
    int idx = blockIdx.x * blockDim.x + threadIdx.x;   // 512*512
    int i = idx >> 9, j = idx & 511;
    float v = Gaug[(size_t)i * 1024 + 512 + j] / Gaug[(size_t)i * 1024 + i];
    X[(size_t)i * 512 + j]    = v;   // N0 = K^T
    Kout[(size_t)j * 512 + i] = v;   // K row-major
}

// ---------------------------------------------------------------------------
// Per-column L2 norms of (Zp - Zt), one wave per column (wave32).
// ---------------------------------------------------------------------------
__global__ __launch_bounds__(256)
void k_colnorm(const float* __restrict__ Zp, const float* __restrict__ Zt,
               float* __restrict__ norms) {
    int w = threadIdx.x >> 5, lane = threadIdx.x & 31;
    int n = blockIdx.x * 8 + w;
    const float* a = Zp + (size_t)n * 512;
    const float* b = Zt + (size_t)n * 512;
    float s = 0.0f;
    for (int i = lane; i < 512; i += 32) {
        float d = a[i] - b[i];
        s += d * d;
    }
    #pragma unroll
    for (int off = 16; off > 0; off >>= 1) s += __shfl_down(s, off, 32);
    if (lane == 0) norms[n] = sqrtf(s);
}

// Deterministic single-block mean of 8192 norms.
__global__ __launch_bounds__(256)
void k_mean(const float* __restrict__ norms, float* __restrict__ out) {
    __shared__ float sm[256];
    float s = 0.0f;
    for (int i = threadIdx.x; i < 8192; i += 256) s += norms[i];
    sm[threadIdx.x] = s;
    __syncthreads();
    for (int o = 128; o > 0; o >>= 1) {
        if ((int)threadIdx.x < o) sm[threadIdx.x] += sm[threadIdx.x + o];
        __syncthreads();
    }
    if (threadIdx.x == 0) out[0] = sm[0] * (1.0f / 8192.0f);
}

// ---------------------------------------------------------------------------
// Host orchestration
// ---------------------------------------------------------------------------
extern "C" void kernel_launch(void* const* d_in, const int* in_sizes, int n_in,
                              void* d_out, int out_size, void* d_ws, size_t ws_size,
                              hipStream_t stream) {
    (void)in_sizes; (void)n_in; (void)out_size; (void)ws_size;
    const float* X  = (const float*)d_in[1];
    const float* W1 = (const float*)d_in[2];
    const float* b1 = (const float*)d_in[3];
    const float* W2 = (const float*)d_in[4];
    const float* b2 = (const float*)d_in[5];
    const float* W3 = (const float*)d_in[6];
    const float* b3 = (const float*)d_in[7];
    const float* W4 = (const float*)d_in[8];
    const float* b4 = (const float*)d_in[9];
    float* out = (float*)d_out;

    const int M = 8192, D = 4096, L = 512;

    char* ws = (char*)d_ws;
    size_t off = 0;
    auto alloc = [&](size_t bytes) -> void* {
        void* p = ws + off;
        off += (bytes + 255) & ~(size_t)255;
        return p;
    };
    unsigned short* Wt1 = (unsigned short*)alloc((size_t)L * D * 2);
    unsigned short* Wt2 = (unsigned short*)alloc((size_t)L * L * 2);
    unsigned short* Wt3 = (unsigned short*)alloc((size_t)L * L * 2);
    unsigned short* Wt4 = (unsigned short*)alloc((size_t)L * L * 2);
    unsigned short* H1  = (unsigned short*)alloc((size_t)M * L * 2);
    unsigned short* H2  = (unsigned short*)alloc((size_t)M * L * 2);
    float* Zt    = (float*)alloc((size_t)M * L * 4);   // (h@W4+b4), rows = samples
    float* Gaug  = (float*)alloc((size_t)L * 1024 * 4);
    float* Nb0   = (float*)alloc((size_t)L * L * 4);
    float* Nb1   = (float*)alloc((size_t)L * L * 4);
    float* Zp    = (float*)alloc((size_t)M * L * 4);   // Z_pred^T, rows = steps
    float* norms = (float*)alloc((size_t)M * 4);

    // 1) pack weights transposed to bf16
    k_pack_wt<<<(D * L + 255) / 256, 256, 0, stream>>>(W1, Wt1, D, L);
    k_pack_wt<<<(L * L + 255) / 256, 256, 0, stream>>>(W2, Wt2, L, L);
    k_pack_wt<<<(L * L + 255) / 256, 256, 0, stream>>>(W3, Wt3, L, L);
    k_pack_wt<<<(L * L + 255) / 256, 256, 0, stream>>>(W4, Wt4, L, L);

    // 2) MLP layers (bf16 WMMA, f32 accumulate)
    dim3 gL(L / 128, M / 128);
    k_layer_gemm<<<gL, 256, 0, stream>>>(X,  1, Wt1, b1, H1, nullptr, 0, M, D);
    k_layer_gemm<<<gL, 256, 0, stream>>>(H1, 0, Wt2, b2, H2, nullptr, 0, M, L);
    k_layer_gemm<<<gL, 256, 0, stream>>>(H2, 0, Wt3, b3, H1, nullptr, 0, M, L);
    k_layer_gemm<<<gL, 256, 0, stream>>>(H1, 0, Wt4, b4, nullptr, Zt, 1, M, L);

    // 3) Gaug = [ Z1 Z1^T + L*ncoef*I  |  Z1 Z2^T ]  (= [G | A2^T])
    dim3 gT(4, 32);
    k_gemm_f32_tn<<<gT, 256, 0, stream>>>(Zt, Zt,              Gaug, M / 2, 1024, 0,   (float)L * 0.0001f);
    k_gemm_f32_tn<<<gT, 256, 0, stream>>>(Zt, Zt + (size_t)(M / 2) * L, Gaug, M / 2, 1024, 512, 0.0f);

    // 4) solve G X = A2^T  (X = K^T)
    for (int p = 0; p < L; ++p)
        k_gj_elim<<<(L * 1024) / 256, 256, 0, stream>>>(Gaug, p);
    k_gj_final<<<(L * L) / 256, 256, 0, stream>>>(Gaug, Nb0, out + 1);

    // 5) scan -> log-depth doubling:  Zp rows [2^k,2^{k+1}) = Zp[0:2^k] @ N_k,
    //    N_{k+1} = N_k @ N_k   (N_0 = K^T)
    hipMemcpyAsync(Zp, Zt, (size_t)L * 4, hipMemcpyDeviceToDevice, stream); // z0 = Zt row 0
    float* Ncur = Nb0;
    float* Nnxt = Nb1;
    for (int k = 0; k < 13; ++k) {
        int m = 1 << k;
        dim3 gB(L / 128, (m + 15) / 16);
        k_gemm_f32_nn<<<gB, 256, 0, stream>>>(Zp, Ncur, Zp + (size_t)m * L, m, L, L, L);
        if (k < 12) {
            k_gemm_f32_nn<<<dim3(L / 128, L / 16), 256, 0, stream>>>(Ncur, Ncur, Nnxt, L, L, L, L);
            float* t = Ncur; Ncur = Nnxt; Nnxt = t;
        }
    }

    // 6) mse = mean_n || Zp[n] - Zt[n] ||_2   (deterministic reduction)
    k_colnorm<<<M / 8, 256, 0, stream>>>(Zp, Zt, norms);
    k_mean<<<1, 256, 0, stream>>>(norms, out);
}
